// GMDecoder_19361712570910
// MI455X (gfx1250) — matrix operand order repaired and checked by reading, not compiled, not run on hardware
//
#include <hip/hip_runtime.h>
#include <hip/hip_bf16.h>
#include <math.h>

typedef _Float16 v4h  __attribute__((ext_vector_type(4)));
typedef _Float16 v8h  __attribute__((ext_vector_type(8)));
typedef _Float16 v16h __attribute__((ext_vector_type(16)));
typedef float    v8f  __attribute__((ext_vector_type(8)));
typedef int      v4i  __attribute__((ext_vector_type(4)));

// --- CDNA5 async global->LDS path (probe-guarded, sync fallback) -----------
#if defined(__has_builtin)
#  if __has_builtin(__builtin_amdgcn_global_load_async_to_lds_b128)
#    define USE_ASYNC_LDS 1
#  endif
#endif
#ifndef USE_ASYNC_LDS
#  define USE_ASYNC_LDS 0
#endif

__device__ __forceinline__ void copy16_g2l(const _Float16* gsrc, v8h* ldst)
{
#if USE_ASYNC_LDS
    typedef __attribute__((address_space(1))) v4i GV;   // global int4
    typedef __attribute__((address_space(3))) v4i LV;   // LDS int4
    __builtin_amdgcn_global_load_async_to_lds_b128((GV*)gsrc, (LV*)ldst, 0, 0);
#else
    *ldst = *(const v8h*)gsrc;
#endif
}

__device__ __forceinline__ void wait_g2l()
{
#if USE_ASYNC_LDS
#  if __has_builtin(__builtin_amdgcn_s_wait_asynccnt)
    __builtin_amdgcn_s_wait_asynccnt(0);
#  else
    asm volatile("s_wait_asynccnt 0" ::: "memory");
#  endif
#endif
}

// ---------------------------------------------------------------------------
// head: mean = y@Wm.T + bm ; cov = sqrt(softplus(y@Wc.T + bc))
// ---------------------------------------------------------------------------
__global__ __launch_bounds__(256)
void head_kernel(const float* __restrict__ y,
                 const float* __restrict__ Wm, const float* __restrict__ bm,
                 const float* __restrict__ Wc, const float* __restrict__ bc,
                 float* __restrict__ mean, float* __restrict__ cov,
                 int d0, int Kin)
{
    int i = blockIdx.x * blockDim.x + threadIdx.x;
    if (i >= d0) return;
    float sm = bm[i], sc = bc[i];
    const float* wm = Wm + (size_t)i * Kin;
    const float* wc = Wc + (size_t)i * Kin;
    for (int k = 0; k < Kin; k += 4) {
        float4 yv = *(const float4*)(y + k);
        float4 a  = *(const float4*)(wm + k);
        float4 b  = *(const float4*)(wc + k);
        sm += yv.x * a.x + yv.y * a.y + yv.z * a.z + yv.w * a.w;
        sc += yv.x * b.x + yv.y * b.y + yv.z * b.z + yv.w * b.w;
    }
    float sp = (sc > 20.0f) ? sc : log1pf(expf(sc));   // softplus, overflow-safe
    mean[i] = sm;
    cov[i]  = sqrtf(sp);
}

// ---------------------------------------------------------------------------
// h0 = relu(mean + z * cov)  -> f16
// ---------------------------------------------------------------------------
__global__ __launch_bounds__(256)
void h0_kernel(const float* __restrict__ z,
               const float* __restrict__ mean, const float* __restrict__ cov,
               _Float16* __restrict__ h0, int total, int d0)
{
    int i = (blockIdx.x * blockDim.x + threadIdx.x) * 4;
    if (i >= total) return;
    int c = i % d0;
    float4 zv = *(const float4*)(z + i);
    float r0 = mean[c + 0] + zv.x * cov[c + 0];
    float r1 = mean[c + 1] + zv.y * cov[c + 1];
    float r2 = mean[c + 2] + zv.z * cov[c + 2];
    float r3 = mean[c + 3] + zv.w * cov[c + 3];
    r0 = r0 > 0.f ? r0 : 0.f;  r1 = r1 > 0.f ? r1 : 0.f;
    r2 = r2 > 0.f ? r2 : 0.f;  r3 = r3 > 0.f ? r3 : 0.f;
    v4h o = { (_Float16)r0, (_Float16)r1, (_Float16)r2, (_Float16)r3 };
    *(v4h*)(h0 + i) = o;
}

// ---------------------------------------------------------------------------
// stream convert f32 -> f16
// ---------------------------------------------------------------------------
__global__ __launch_bounds__(256)
void cvt_f32_f16(const float* __restrict__ src, _Float16* __restrict__ dst, int n)
{
    int i = (blockIdx.x * blockDim.x + threadIdx.x) * 4;
    if (i >= n) return;
    float4 v = *(const float4*)(src + i);
    v4h o = { (_Float16)v.x, (_Float16)v.y, (_Float16)v.z, (_Float16)v.w };
    *(v4h*)(dst + i) = o;
}

// ---------------------------------------------------------------------------
// WMMA GEMM:  out[M,N] = act( A[M,KD] @ W[ND,KD]^T + bias[ND] )
// BM=128, BN=256, BK=32. 256 threads = 8 wave32 in a 2x4 layout.
// Each wave: 64x64 of C = 4x4 tiles of v_wmma_f32_16x16x32_f16.
// Double-buffered LDS (60KB), rows padded to 40 halves -> conflict-free b128.
// Next k-block streamed global->LDS (async on CDNA5) under the WMMAs.
// ---------------------------------------------------------------------------
template<int KD, int ND, bool RELU, bool OUT_F16>
__global__ __launch_bounds__(256)
void gemm_wmma(const _Float16* __restrict__ A,   // [M,KD]
               const _Float16* __restrict__ W,   // [ND,KD]
               const float*    __restrict__ bias,// [ND]
               void*           __restrict__ out, // [M,ND] f16 or f32
               int M)
{
    constexpr int BM = 128, BN = 256, BK = 32;
    constexpr int RP = 5;                    // v8h per padded row (32+8 halves)
    __shared__ v8h lsA[2][BM * RP];          // 20 KB
    __shared__ v8h lsB[2][BN * RP];          // 40 KB

    const int tid    = threadIdx.x;
    const int lane   = tid & 31;
    const int wid    = tid >> 5;
    const int wave_m = wid >> 2;             // 0..1
    const int wave_n = wid & 3;              // 0..3
    const int lrow   = lane & 15;
    const int hi     = lane >> 4;            // lane half select

    const int m0 = blockIdx.y * BM;
    const int n0 = blockIdx.x * BN;

    // staging map: 4 threads per row (16B each), 64 rows per pass
    const int srow = tid >> 2;               // 0..63
    const int scol = tid & 3;                // 0..3

    v8f acc[4][4] = {};

    const int nk = KD / BK;

    // issue tile copy for k-block `k0` into buffer `buf`
    auto copy_tile = [&](int buf, int k0) {
#pragma unroll
        for (int p = 0; p < 2; ++p)          // A: 128 rows
            copy16_g2l(A + (size_t)(m0 + srow + p * 64) * KD + k0 + scol * 8,
                       &lsA[buf][(srow + p * 64) * RP + scol]);
#pragma unroll
        for (int p = 0; p < 4; ++p)          // B: 256 rows
            copy16_g2l(W + (size_t)(n0 + srow + p * 64) * KD + k0 + scol * 8,
                       &lsB[buf][(srow + p * 64) * RP + scol]);
    };

    copy_tile(0, 0);
    wait_g2l();
    __syncthreads();

    for (int kb = 0; kb < nk; ++kb) {
        const int  cur  = kb & 1;
        const int  nxt  = cur ^ 1;
        const bool more = (kb + 1) < nk;

        if (more) copy_tile(nxt, (kb + 1) * BK);   // streams while WMMAs run

        // A fragments: lanes 0-15 row M=lrow get K[0..7 | 16..23],
        //              lanes 16-31 same rows get K[8..15 | 24..31]
        v16h afrag[4];
#pragma unroll
        for (int mt = 0; mt < 4; ++mt) {
            int r = wave_m * 64 + mt * 16 + lrow;
            v8h x0 = lsA[cur][r * RP + hi];
            v8h x1 = lsA[cur][r * RP + hi + 2];
            afrag[mt] = __builtin_shufflevector(x0, x1, 0,1,2,3,4,5,6,7,8,9,10,11,12,13,14,15);
        }
        // B fragments: lane holds column n=lrow, K[hi*8..hi*8+7 | 16+hi*8..16+hi*8+7]
        v16h bfrag[4];
#pragma unroll
        for (int nt = 0; nt < 4; ++nt) {
            int c = wave_n * 64 + nt * 16 + lrow;
            v8h x0 = lsB[cur][c * RP + hi];
            v8h x1 = lsB[cur][c * RP + hi + 2];
            bfrag[nt] = __builtin_shufflevector(x0, x1, 0,1,2,3,4,5,6,7,8,9,10,11,12,13,14,15);
        }

#pragma unroll
        for (int mt = 0; mt < 4; ++mt)
#pragma unroll
            for (int nt = 0; nt < 4; ++nt)
                acc[mt][nt] = __builtin_amdgcn_wmma_f32_16x16x32_f16(
                    false, afrag[mt], false, bfrag[nt],
                    (short)0, acc[mt][nt], false, false);

        if (more) wait_g2l();
        __syncthreads();
    }

    // epilogue: C layout -> VGPR r holds row r (lanes 0-15) / row 8+r (lanes 16-31)
    _Float16* outh = (_Float16*)out;
    float*    outf = (float*)out;
#pragma unroll
    for (int mt = 0; mt < 4; ++mt) {
#pragma unroll
        for (int nt = 0; nt < 4; ++nt) {
            int gn      = n0 + wave_n * 64 + nt * 16 + lrow;
            int gm_base = m0 + wave_m * 64 + mt * 16 + hi * 8;
            float bv = bias[gn];
#pragma unroll
            for (int r = 0; r < 8; ++r) {
                float v = acc[mt][nt][r] + bv;
                if (RELU) v = v > 0.f ? v : 0.f;
                size_t off = (size_t)(gm_base + r) * ND + gn;
                if (OUT_F16) outh[off] = (_Float16)v;
                else         outf[off] = v;
            }
        }
    }
}

// ---------------------------------------------------------------------------
// launch
// ---------------------------------------------------------------------------
extern "C" void kernel_launch(void* const* d_in, const int* in_sizes, int n_in,
                              void* d_out, int out_size, void* d_ws, size_t ws_size,
                              hipStream_t stream)
{
    constexpr int Kin = 128;
    constexpr int d0 = 2048, d1 = 4096, d2 = 4096, d3 = 2048;
    constexpr int Bm = 4096;

    const float* y  = (const float*)d_in[0];
    const float* z  = (const float*)d_in[1];
    const float* Wm = (const float*)d_in[2];
    const float* bm = (const float*)d_in[3];
    const float* Wc = (const float*)d_in[4];
    const float* bc = (const float*)d_in[5];
    const float* W1 = (const float*)d_in[6];
    const float* b1 = (const float*)d_in[7];
    const float* W2 = (const float*)d_in[8];
    const float* b2 = (const float*)d_in[9];
    const float* W3 = (const float*)d_in[10];
    const float* b3 = (const float*)d_in[11];

    // workspace layout (256B aligned)
    char* ws = (char*)d_ws;
    constexpr size_t ALN = 256;
    auto align = [](size_t x) { return (x + ALN - 1) & ~(ALN - 1); };
    size_t off = 0;
    float*    mean = (float*)(ws + off);    off = align(off + (size_t)d0 * 4);
    float*    cov  = (float*)(ws + off);    off = align(off + (size_t)d0 * 4);
    _Float16* h0   = (_Float16*)(ws + off); off = align(off + (size_t)Bm * d0 * 2);
    _Float16* h1   = (_Float16*)(ws + off); off = align(off + (size_t)Bm * d1 * 2);
    _Float16* h2   = (_Float16*)(ws + off); off = align(off + (size_t)Bm * d2 * 2);
    _Float16* W1h  = (_Float16*)(ws + off); off = align(off + (size_t)d1 * d0 * 2);
    _Float16* W2h  = (_Float16*)(ws + off); off = align(off + (size_t)d2 * d1 * 2);
    _Float16* W3h  = (_Float16*)(ws + off); off = align(off + (size_t)d3 * d2 * 2);
    (void)ws_size;

    // 1) head: mean / cov_diag
    head_kernel<<<d0 / 256, 256, 0, stream>>>(y, Wm, bm, Wc, bc, mean, cov, d0, Kin);

    // 2) weight conversion f32 -> f16 (must run every call; no caching)
    {
        int n1 = d1 * d0, n2 = d2 * d1, n3 = d3 * d2;
        cvt_f32_f16<<<(n1 / 4 + 255) / 256, 256, 0, stream>>>(W1, W1h, n1);
        cvt_f32_f16<<<(n2 / 4 + 255) / 256, 256, 0, stream>>>(W2, W2h, n2);
        cvt_f32_f16<<<(n3 / 4 + 255) / 256, 256, 0, stream>>>(W3, W3h, n3);
    }

    // 3) h0 = relu(mean + z*cov) -> f16
    {
        int total = Bm * d0;
        h0_kernel<<<(total / 4 + 255) / 256, 256, 0, stream>>>(z, mean, cov, h0, total, d0);
    }

    // 4) three WMMA GEMMs (N-tile = 256, M-tile = 128)
    gemm_wmma<d0, d1, true,  true ><<<dim3(d1 / 256, Bm / 128), 256, 0, stream>>>(
        h0, W1h, b1, (void*)h1, Bm);
    gemm_wmma<d1, d2, true,  true ><<<dim3(d2 / 256, Bm / 128), 256, 0, stream>>>(
        h1, W2h, b2, (void*)h2, Bm);
    gemm_wmma<d2, d3, false, false><<<dim3(d3 / 256, Bm / 128), 256, 0, stream>>>(
        h2, W3h, b3, d_out, Bm);
}
